// DrawModelSN_86895778332704
// MI455X (gfx1250) — compile-verified, matrix operand order checked by reading, not run on hardware
//
#include <hip/hip_runtime.h>
#include <math.h>

// ---------------- model dims ----------------
#define T_STEPS 16
#define A_DIM 128
#define B_DIM 128
#define N_ATT 32
#define Z_DIM 512
#define HID 2048           // ENC == DEC == 2048
#define BATCH 512
#define IN_ENC 4096        // 2*N*N + DEC
#define IMG (A_DIM*B_DIM)  // 16384
#define LOG2PI 1.8378770664093453f

typedef __attribute__((ext_vector_type(16))) __bf16 v16bf;
typedef __attribute__((ext_vector_type(8)))  float v8f;
typedef __attribute__((ext_vector_type(8)))  unsigned short u16x8;
typedef __attribute__((ext_vector_type(4)))  unsigned int v4u;
typedef __attribute__((ext_vector_type(8)))  int v8i;
typedef __attribute__((ext_vector_type(4)))  int v4i;

#if __has_builtin(__builtin_amdgcn_tensor_load_to_lds)
#define HAVE_TDM 1
#else
#define HAVE_TDM 0
#endif

__device__ __forceinline__ unsigned short f32_bf16(float f) {
  unsigned u = __float_as_uint(f);
  u += 0x7FFFu + ((u >> 16) & 1u);       // round-to-nearest-even
  return (unsigned short)(u >> 16);
}
__device__ __forceinline__ float sigmoidf(float x) { return 1.0f / (1.0f + expf(-x)); }

#if HAVE_TDM
// Issue a TDM 2-D tile load: tensor is (rows x K) bf16 row-major, tile is
// tile_d1 rows x tile_d0 cols starting at element offset elem_off.
__device__ __forceinline__ void tdm_load_2d(unsigned lds_off, const unsigned short* gbase,
                                            long elem_off, int K, int rows,
                                            int tile_d0, int tile_d1)
{
  unsigned long long ga = (unsigned long long)(uintptr_t)gbase + (unsigned long long)elem_off * 2ull;
  v4u g0;
  g0[0] = 1u;                                              // count=1, user mode
  g0[1] = lds_off;                                         // LDS byte address
  g0[2] = (unsigned)(ga & 0xFFFFFFFFu);                    // global_addr[31:0]
  g0[3] = (unsigned)((ga >> 32) & 0x01FFFFFFu) | (2u << 30); // global_addr[56:32] | type=2
  unsigned td0 = (unsigned)K, td1 = (unsigned)rows;
  v8i g1;
  g1[0] = (int)(1u << 16);                                 // data_size=1 (2 bytes)
  g1[1] = (int)((td0 & 0xFFFFu) << 16);                    // tensor_dim0[15:0] @ bits63:48
  g1[2] = (int)(((td0 >> 16) & 0xFFFFu) | ((td1 & 0xFFFFu) << 16));
  g1[3] = (int)(((td1 >> 16) & 0xFFFFu) | (((unsigned)tile_d0 & 0xFFFFu) << 16));
  g1[4] = (int)((unsigned)tile_d1 & 0xFFFFu);              // tile_dim1; tile_dim2=0
  g1[5] = (int)td0;                                        // tensor_dim0_stride[31:0] = K
  g1[6] = 0;
  g1[7] = 0;
  v4i g2 = {0, 0, 0, 0};
  v4i g3 = {0, 0, 0, 0};
#if defined(__clang_major__) && __clang_major__ >= 23
  v8i g4 = {0, 0, 0, 0, 0, 0, 0, 0};
  __builtin_amdgcn_tensor_load_to_lds(g0, g1, g2, g3, g4, 0);
#else
  __builtin_amdgcn_tensor_load_to_lds(g0, g1, g2, g3, 0);
#endif
}
#endif

__device__ __forceinline__ unsigned lds_off_of(const void* p) {
  // generic LDS address: offset lives in the low 32 bits
  return (unsigned)(uintptr_t)p;
}

// ---------------- utility kernels ----------------
__global__ void k_zero(float* __restrict__ p, long n) {
  long i = (long)blockIdx.x * blockDim.x + threadIdx.x;
  long s = (long)gridDim.x * blockDim.x;
  for (; i < n; i += s) p[i] = 0.0f;
}

__global__ void k_cvt_bf16(const float* __restrict__ src, unsigned short* __restrict__ dst, long n) {
  long i = (long)blockIdx.x * blockDim.x + threadIdx.x;
  long s = (long)gridDim.x * blockDim.x;
  for (; i < n; i += s) dst[i] = f32_bf16(src[i]);
}

__global__ void k_xhat(const float* __restrict__ x, const float* __restrict__ canvas,
                       float* __restrict__ xhat, long n) {
  long i = (long)blockIdx.x * blockDim.x + threadIdx.x;
  long s = (long)gridDim.x * blockDim.x;
  for (; i < n; i += s) xhat[i] = x[i] - sigmoidf(canvas[i]);
}

// ---------------- WMMA GEMM:  C[M,N] = (acc ? C : bias) + A(bf16, MxK) * W(bf16, NxK)^T --------
// Block tile: 128(M) x 64(N), K chunk 64, double-buffered TDM staging.
__global__ __launch_bounds__(256) void k_wmma_gemm(
    const unsigned short* __restrict__ A, const unsigned short* __restrict__ W,
    const float* __restrict__ bias, float* __restrict__ C,
    int M, int N, int K, int accumulate)
{
  __shared__ unsigned short sA[2][128 * 64];
  __shared__ unsigned short sW[2][64 * 64];

  const int tid  = threadIdx.x;
  const int lane = tid & 31;
  const int wave = tid >> 5;           // 0..7, wave owns 16-row slab
  const int bm = blockIdx.y * 128;
  const int bn = blockIdx.x * 64;
  const int mrow = lane & 15;
  const int kb   = (lane >> 4) * 8;    // wave32 fragment K-base split

  v8f acc[4] = {};

#if HAVE_TDM
  if (wave == 0)      tdm_load_2d(lds_off_of(&sW[0][0]), W, (long)bn * K, K, N, 64, 64);
  else if (wave == 1) tdm_load_2d(lds_off_of(&sA[0][0]), A, (long)bm * K, K, M, 64, 128);
#endif

  int buf = 0;
  for (int k0 = 0; k0 < K; k0 += 64, buf ^= 1) {
    const bool has_next = (k0 + 64) < K;
#if HAVE_TDM
    if (has_next) {
      if (wave == 0)      tdm_load_2d(lds_off_of(&sW[buf ^ 1][0]), W, (long)bn * K + (k0 + 64), K, N, 64, 64);
      else if (wave == 1) tdm_load_2d(lds_off_of(&sA[buf ^ 1][0]), A, (long)bm * K + (k0 + 64), K, M, 64, 128);
    }
    if (has_next) __builtin_amdgcn_s_wait_tensorcnt(1);
    else          __builtin_amdgcn_s_wait_tensorcnt(0);
#else
    (void)has_next;
    // cooperative fallback staging: 256 threads copy both tiles
    for (int idx = tid; idx < 128 * 8; idx += 256) {          // A: 128 rows x 64 cols, 8 halves/txn
      int r = idx >> 3, c = (idx & 7) * 8;
      *(u16x8*)&sA[buf][r * 64 + c] = *(const u16x8*)(A + (long)(bm + r) * K + k0 + c);
    }
    for (int idx = tid; idx < 64 * 8; idx += 256) {           // W: 64 rows x 64 cols
      int r = idx >> 3, c = (idx & 7) * 8;
      *(u16x8*)&sW[buf][r * 64 + c] = *(const u16x8*)(W + (long)(bn + r) * K + k0 + c);
    }
#endif
    __syncthreads();

    union FragU { v16bf v; u16x8 h[2]; };
    const unsigned short* a0 = &sA[buf][(wave * 16 + mrow) * 64];
    #pragma unroll
    for (int kc = 0; kc < 2; ++kc) {
      FragU fa;
      fa.h[0] = *(const u16x8*)(a0 + kc * 32 + kb);
      fa.h[1] = *(const u16x8*)(a0 + kc * 32 + 16 + kb);
      #pragma unroll
      for (int t = 0; t < 4; ++t) {
        const unsigned short* bp = &sW[buf][(t * 16 + mrow) * 64 + kc * 32];
        FragU fb;
        fb.h[0] = *(const u16x8*)(bp + kb);
        fb.h[1] = *(const u16x8*)(bp + 16 + kb);
        acc[t] = __builtin_amdgcn_wmma_f32_16x16x32_bf16(false, fa.v, false, fb.v,
                                                         (short)0, acc[t], false, false);
      }
    }
    __syncthreads();
  }

  // epilogue: C layout VGPR r -> M = r + 8*(lane>>4), N = lane&15
  const int crow0 = bm + wave * 16 + 8 * (lane >> 4);
  #pragma unroll
  for (int t = 0; t < 4; ++t) {
    const int ccol = bn + t * 16 + (lane & 15);
    #pragma unroll
    for (int r = 0; r < 8; ++r) {
      long i0 = (long)(crow0 + r) * N + ccol;
      float v = acc[t][r];
      if (accumulate) v += C[i0];
      else if (bias)  v += bias[ccol];
      C[i0] = v;
    }
  }
}

// ---------------- LSTM pointwise (gate order i,f,g,o) ----------------
__global__ void k_lstm(const float* __restrict__ g, float* __restrict__ h,
                       float* __restrict__ c, long n /* batch*H */, int H) {
  long i = (long)blockIdx.x * blockDim.x + threadIdx.x;
  long s = (long)gridDim.x * blockDim.x;
  for (; i < n; i += s) {
    long b = i / H; int j = (int)(i % H);
    const float* gr = g + b * (long)(4 * H);
    float gi = gr[j], gf = gr[H + j], gg = gr[2 * H + j], go = gr[3 * H + j];
    float cn = sigmoidf(gf) * c[i] + sigmoidf(gi) * tanhf(gg);
    c[i] = cn;
    h[i] = sigmoidf(go) * tanhf(cn);
  }
}

// ---------------- MDL renorm: stats, logsumexp, scale(+bf16 mirror) ----------------
__global__ __launch_bounds__(1024) void k_mdl_stats(const float* __restrict__ h, long n,
                                                    float* __restrict__ scal) {
  __shared__ float s1[1024], s2[1024];
  float a = 0.f, b = 0.f;
  for (long i = threadIdx.x; i < n; i += 1024) { float v = h[i]; a += v; b += v * v; }
  s1[threadIdx.x] = a; s2[threadIdx.x] = b; __syncthreads();
  for (int st = 512; st > 0; st >>= 1) {
    if ((int)threadIdx.x < st) { s1[threadIdx.x] += s1[threadIdx.x + st]; s2[threadIdx.x] += s2[threadIdx.x + st]; }
    __syncthreads();
  }
  if (threadIdx.x == 0) {
    float mean = s1[0] / (float)n;
    float var  = (s2[0] - (float)n * mean * mean) / (float)(n - 1);
    scal[0] = mean;
    scal[1] = sqrtf(fmaxf(var, 1e-30f));
  }
}

__global__ __launch_bounds__(1024) void k_mdl_lse(const float* __restrict__ h, long n,
                                                  float* __restrict__ scal) {
  __shared__ float s1[1024];
  float mean = scal[0], std = scal[1];
  float lstd = logf(std);
  float a = 0.f;
  for (long i = threadIdx.x; i < n; i += 1024) {
    float t = (h[i] - mean) / std;
    a += expf(-0.5f * t * t - lstd - 0.5f * LOG2PI);
  }
  s1[threadIdx.x] = a; __syncthreads();
  for (int st = 512; st > 0; st >>= 1) {
    if ((int)threadIdx.x < st) s1[threadIdx.x] += s1[threadIdx.x + st];
    __syncthreads();
  }
  if (threadIdx.x == 0) {
    float lse = logf(s1[0]);
    float l = (lse > 0.f) ? log1pf(expf(-lse)) : (log1pf(expf(lse)) - lse); // logaddexp(0,lse)-lse
    scal[2] = l;
  }
}

__global__ void k_scale_bf(float* __restrict__ h, const float* __restrict__ scal,
                           unsigned short* __restrict__ hbf, long n) {
  float l = scal[2];
  long i = (long)blockIdx.x * blockDim.x + threadIdx.x;
  long s = (long)gridDim.x * blockDim.x;
  for (; i < n; i += s) { float v = h[i] * l; h[i] = v; hbf[i] = f32_bf16(v); }
}

// ---------------- sampleQ:  z = mu + exp(sp) * eps_t  (bf16 out, GEMM-only consumer) -------
__global__ void k_sampleq(const float* __restrict__ mu, const float* __restrict__ sp,
                          const float* __restrict__ eps_t, unsigned short* __restrict__ zbf, long n) {
  long i = (long)blockIdx.x * blockDim.x + threadIdx.x;
  long s = (long)gridDim.x * blockDim.x;
  for (; i < n; i += s) zbf[i] = f32_bf16(mu[i] + expf(sp[i]) * eps_t[i]);
}

// ---------------- attention filterbank (per batch element) ----------------
__global__ __launch_bounds__(128) void k_attn(const float* __restrict__ hdec,
    const float* __restrict__ attn_W, const float* __restrict__ attn_b,
    float* __restrict__ Fx, float* __restrict__ Fy, float* __restrict__ gamma)
{
  const int b = blockIdx.x, tid = threadIdx.x;
  __shared__ float red[128];
  __shared__ float p[5];
  __shared__ float gx, gy, sigma2, delta;
  const float* hb = hdec + (long)b * HID;
  for (int j = 0; j < 5; ++j) {
    float s = 0.f;
    for (int k = tid; k < HID; k += 128) s += hb[k] * attn_W[j * HID + k];
    red[tid] = s; __syncthreads();
    for (int st = 64; st > 0; st >>= 1) { if (tid < st) red[tid] += red[tid + st]; __syncthreads(); }
    if (tid == 0) p[j] = red[0] + attn_b[j];
    __syncthreads();
  }
  if (tid == 0) {
    gx = 0.5f * (A_DIM + 1) * (p[0] + 1.f);
    gy = 0.5f * (B_DIM + 1) * (p[1] + 1.f);
    sigma2 = expf(p[2]);
    delta  = ((float)(A_DIM - 1) / (float)(N_ATT - 1)) * expf(p[3]);
    gamma[b] = expf(p[4]);
  }
  __syncthreads();
  for (int n = 0; n < N_ATT; ++n) {
    float rng = (float)n - (float)N_ATT * 0.5f - 0.5f;      // n - 16.5
    float mux = gx + rng * delta;
    float t = ((float)tid - mux) / (2.f * sigma2);
    float e = expf(-t * t);
    red[tid] = e; __syncthreads();
    for (int st = 64; st > 0; st >>= 1) { if (tid < st) red[tid] += red[tid + st]; __syncthreads(); }
    Fx[((long)b * N_ATT + n) * A_DIM + tid] = e / (red[0] + 1e-9f);
    __syncthreads();
    float muy = gy + rng * delta;
    float t2 = ((float)tid - muy) / (2.f * sigma2);
    float e2 = expf(-t2 * t2);
    red[tid] = e2; __syncthreads();
    for (int st = 64; st > 0; st >>= 1) { if (tid < st) red[tid] += red[tid + st]; __syncthreads(); }
    Fy[((long)b * N_ATT + n) * B_DIM + tid] = e2 / (red[0] + 1e-9f);
    __syncthreads();
  }
}

// ---------------- read glimpses + pack encoder input xcat = [r, h_dec] (bf16) ---------------
__global__ __launch_bounds__(256) void k_read(const float* __restrict__ x,
    const float* __restrict__ xhat, const float* __restrict__ Fx,
    const float* __restrict__ Fy, const float* __restrict__ gamma,
    const float* __restrict__ hdec, unsigned short* __restrict__ xcatbf)
{
  const int b = blockIdx.x, tid = threadIdx.x;
  __shared__ float tmp[N_ATT * A_DIM];   // 32x128
  const float* fx = Fx + (long)b * (N_ATT * A_DIM);
  const float* fy = Fy + (long)b * (N_ATT * B_DIM);
  const float gam = gamma[b];
  unsigned short* rb = xcatbf + (long)b * IN_ENC;
  for (int pass = 0; pass < 2; ++pass) {
    const float* img = (pass ? xhat : x) + (long)b * IMG;   // (B,A) row-major
    for (int idx = tid; idx < N_ATT * A_DIM; idx += 256) {
      int n = idx >> 7, a = idx & 127;
      float s = 0.f;
      for (int Bi = 0; Bi < B_DIM; ++Bi) s += fy[n * B_DIM + Bi] * img[Bi * A_DIM + a];
      tmp[idx] = s;
    }
    __syncthreads();
    for (int idx = tid; idx < N_ATT * N_ATT; idx += 256) {
      int n = idx >> 5, m = idx & 31;
      float s = 0.f;
      for (int a = 0; a < A_DIM; ++a) s += tmp[n * A_DIM + a] * fx[m * A_DIM + a];
      rb[pass * (N_ATT * N_ATT) + idx] = f32_bf16(gam * s);
    }
    __syncthreads();
  }
  const float* hb = hdec + (long)b * HID;
  for (int i = tid; i < HID; i += 256) rb[2 * N_ATT * N_ATT + i] = f32_bf16(hb[i]);
}

// ---------------- write: canvas += (Fy^T w Fx) / gamma ----------------
__global__ __launch_bounds__(256) void k_write(const float* __restrict__ wv,
    const float* __restrict__ Fx, const float* __restrict__ Fy,
    const float* __restrict__ gamma, float* __restrict__ canvas)
{
  const int b = blockIdx.x, tid = threadIdx.x;
  __shared__ float sw[N_ATT * N_ATT];
  __shared__ float tmp[N_ATT * A_DIM];
  const float* fx = Fx + (long)b * (N_ATT * A_DIM);
  const float* fy = Fy + (long)b * (N_ATT * B_DIM);
  const float inv_g = 1.0f / gamma[b];
  for (int i = tid; i < N_ATT * N_ATT; i += 256) sw[i] = wv[(long)b * (N_ATT * N_ATT) + i];
  __syncthreads();
  for (int idx = tid; idx < N_ATT * A_DIM; idx += 256) {
    int n = idx >> 7, a = idx & 127;
    float s = 0.f;
    for (int m = 0; m < N_ATT; ++m) s += sw[n * N_ATT + m] * fx[m * A_DIM + a];
    tmp[idx] = s;
  }
  __syncthreads();
  float* cb = canvas + (long)b * IMG;
  for (int idx = tid; idx < IMG; idx += 256) {
    int Bi = idx >> 7, a = idx & 127;
    float s = 0.f;
    for (int n = 0; n < N_ATT; ++n) s += fy[n * B_DIM + Bi] * tmp[n * A_DIM + a];
    cb[idx] += inv_g * s;
  }
}

// ---------------- host driver ----------------
extern "C" void kernel_launch(void* const* d_in, const int* in_sizes, int n_in,
                              void* d_out, int out_size, void* d_ws, size_t ws_size,
                              hipStream_t stream) {
  const float* x       = (const float*)d_in[0];
  const float* eps     = (const float*)d_in[1];
  const float* enc_Wih = (const float*)d_in[2];
  const float* enc_Whh = (const float*)d_in[3];
  const float* enc_b   = (const float*)d_in[4];
  const float* dec_Wih = (const float*)d_in[5];
  const float* dec_Whh = (const float*)d_in[6];
  const float* dec_b   = (const float*)d_in[7];
  const float* mu_W    = (const float*)d_in[8];
  const float* mu_b    = (const float*)d_in[9];
  const float* sig_W   = (const float*)d_in[10];
  const float* sig_b   = (const float*)d_in[11];
  const float* attn_W  = (const float*)d_in[12];
  const float* attn_b  = (const float*)d_in[13];
  const float* w_W     = (const float*)d_in[14];
  const float* w_b     = (const float*)d_in[15];
  float* canvas = (float*)d_out;   // (BATCH, A*B), accumulates in place

  // ---- workspace bump allocator ----
  char* wp = (char*)d_ws;
  auto balloc = [&](size_t bytes) -> void* {
    void* r = (void*)wp;
    wp += (bytes + 255) & ~(size_t)255;
    return r;
  };
  unsigned short* bfEncWih = (unsigned short*)balloc((size_t)4 * HID * IN_ENC * 2);
  unsigned short* bfEncWhh = (unsigned short*)balloc((size_t)4 * HID * HID * 2);
  unsigned short* bfDecWih = (unsigned short*)balloc((size_t)4 * HID * Z_DIM * 2);
  unsigned short* bfDecWhh = (unsigned short*)balloc((size_t)4 * HID * HID * 2);
  unsigned short* bfMuW    = (unsigned short*)balloc((size_t)Z_DIM * HID * 2);
  unsigned short* bfSigW   = (unsigned short*)balloc((size_t)Z_DIM * HID * 2);
  unsigned short* bfWW     = (unsigned short*)balloc((size_t)N_ATT * N_ATT * HID * 2);
  float* states = (float*)balloc((size_t)4 * BATCH * HID * 4); // h_enc,c_enc,h_dec,c_dec
  float* h_enc = states;
  float* c_enc = states + (size_t)BATCH * HID;
  float* h_dec = states + (size_t)2 * BATCH * HID;
  float* c_dec = states + (size_t)3 * BATCH * HID;
  unsigned short* hbf = (unsigned short*)balloc((size_t)2 * BATCH * HID * 2); // bf16 mirrors
  unsigned short* hbf_enc = hbf;
  unsigned short* hbf_dec = hbf + (size_t)BATCH * HID;
  unsigned short* xcatbf = (unsigned short*)balloc((size_t)BATCH * IN_ENC * 2);
  unsigned short* zbf    = (unsigned short*)balloc((size_t)BATCH * Z_DIM * 2);
  float* gbuf  = (float*)balloc((size_t)BATCH * 4 * HID * 4);
  float* mu    = (float*)balloc((size_t)BATCH * Z_DIM * 4);
  float* sp    = (float*)balloc((size_t)BATCH * Z_DIM * 4);
  float* Fx    = (float*)balloc((size_t)BATCH * N_ATT * A_DIM * 4);
  float* Fy    = (float*)balloc((size_t)BATCH * N_ATT * B_DIM * 4);
  float* gam   = (float*)balloc((size_t)BATCH * 4);
  float* wv    = (float*)balloc((size_t)BATCH * N_ATT * N_ATT * 4);
  float* xhat  = (float*)balloc((size_t)BATCH * IMG * 4);
  float* scal  = (float*)balloc(64);

  const dim3 blk256(256);
  const dim3 gsPW(2048);

  auto cvt = [&](const float* s, unsigned short* d, long n) {
    k_cvt_bf16<<<dim3(2048), blk256, 0, stream>>>(s, d, n);
  };
  auto gemm = [&](const unsigned short* Aa, const unsigned short* Wb, const float* bias,
                  float* Cc, int M, int Nn, int K, int acc) {
    k_wmma_gemm<<<dim3(Nn / 64, M / 128), blk256, 0, stream>>>(Aa, Wb, bias, Cc, M, Nn, K, acc);
  };

  // ---- one-time conversions & state init ----
  cvt(enc_Wih, bfEncWih, (long)4 * HID * IN_ENC);
  cvt(enc_Whh, bfEncWhh, (long)4 * HID * HID);
  cvt(dec_Wih, bfDecWih, (long)4 * HID * Z_DIM);
  cvt(dec_Whh, bfDecWhh, (long)4 * HID * HID);
  cvt(mu_W,    bfMuW,    (long)Z_DIM * HID);
  cvt(sig_W,   bfSigW,   (long)Z_DIM * HID);
  cvt(w_W,     bfWW,     (long)N_ATT * N_ATT * HID);
  k_zero<<<gsPW, blk256, 0, stream>>>(states, (long)4 * BATCH * HID);
  k_zero<<<gsPW, blk256, 0, stream>>>((float*)hbf, (long)BATCH * HID); // 2 x 1M ushorts
  k_zero<<<gsPW, blk256, 0, stream>>>(canvas, (long)BATCH * IMG);

  const long nH  = (long)BATCH * HID;
  const long nZ  = (long)BATCH * Z_DIM;
  const long nIm = (long)BATCH * IMG;

  for (int t = 0; t < T_STEPS; ++t) {
    // read attention on current h_dec
    k_attn<<<dim3(BATCH), dim3(128), 0, stream>>>(h_dec, attn_W, attn_b, Fx, Fy, gam);
    k_xhat<<<gsPW, blk256, 0, stream>>>(x, canvas, xhat, nIm);
    k_read<<<dim3(BATCH), blk256, 0, stream>>>(x, xhat, Fx, Fy, gam, h_dec, xcatbf);

    // encoder LSTM
    gemm(xcatbf,  bfEncWih, enc_b,   gbuf, BATCH, 4 * HID, IN_ENC, 0);
    gemm(hbf_enc, bfEncWhh, nullptr, gbuf, BATCH, 4 * HID, HID,    1);
    k_lstm<<<gsPW, blk256, 0, stream>>>(gbuf, h_enc, c_enc, nH, HID);
    k_mdl_stats<<<dim3(1), dim3(1024), 0, stream>>>(h_enc, nH, scal);
    k_mdl_lse<<<dim3(1), dim3(1024), 0, stream>>>(h_enc, nH, scal);
    k_scale_bf<<<gsPW, blk256, 0, stream>>>(h_enc, scal, hbf_enc, nH);

    // sampleQ
    gemm(hbf_enc, bfMuW,  mu_b,  mu, BATCH, Z_DIM, HID, 0);
    gemm(hbf_enc, bfSigW, sig_b, sp, BATCH, Z_DIM, HID, 0);
    k_sampleq<<<gsPW, blk256, 0, stream>>>(mu, sp, eps + (long)t * nZ, zbf, nZ);

    // decoder LSTM (consumes old h_dec mirror before overwrite)
    gemm(zbf,     bfDecWih, dec_b,   gbuf, BATCH, 4 * HID, Z_DIM, 0);
    gemm(hbf_dec, bfDecWhh, nullptr, gbuf, BATCH, 4 * HID, HID,   1);
    k_lstm<<<gsPW, blk256, 0, stream>>>(gbuf, h_dec, c_dec, nH, HID);
    k_mdl_stats<<<dim3(1), dim3(1024), 0, stream>>>(h_dec, nH, scal);
    k_mdl_lse<<<dim3(1), dim3(1024), 0, stream>>>(h_dec, nH, scal);
    k_scale_bf<<<gsPW, blk256, 0, stream>>>(h_dec, scal, hbf_dec, nH);

    // write head
    gemm(hbf_dec, bfWW, w_b, wv, BATCH, N_ATT * N_ATT, HID, 0);
    k_attn<<<dim3(BATCH), dim3(128), 0, stream>>>(h_dec, attn_W, attn_b, Fx, Fy, gam);
    k_write<<<dim3(BATCH), blk256, 0, stream>>>(wv, Fx, Fy, gam, canvas);
  }
  (void)in_sizes; (void)n_in; (void)out_size; (void)ws_size;
}
